// SSIM_74440373174304
// MI455X (gfx1250) — compile-verified, matrix operand order checked by reading, not run on hardware
//
#include <hip/hip_runtime.h>

// ---------------------------------------------------------------------------
// Seq2seq LSTM + attention for MI455X (gfx1250, wave32, WMMA).
// All GEMMs use v_wmma_f32_16x16x32_f16 (f16 in, f32 accum), register-blocked
// 2-4 N-tiles per wave so one A-fragment load feeds 2-4 WMMAs.
// Recurrent parts run as persistent 32-wave workgroups with state in LDS.
// ---------------------------------------------------------------------------

#define B_  64
#define S_  128
#define IN_ 64
#define H_  256
#define D_  256

typedef __attribute__((ext_vector_type(16))) _Float16 v16h;
typedef __attribute__((ext_vector_type(8)))  _Float16 v8h;
typedef __attribute__((ext_vector_type(8)))  float    v8f;

__device__ __forceinline__ float sigf(float x) { return 1.0f / (1.0f + __expf(-x)); }

// Fast tanh: native v_tanh_f32 if available (CDNA5 TRANS op), else 1 trans op.
__device__ __forceinline__ float tanhfast(float x) {
#if __has_builtin(__builtin_amdgcn_tanhf)
  return __builtin_amdgcn_tanhf(x);
#else
  return 1.0f - 2.0f / (__expf(2.0f * x) + 1.0f);
#endif
}

// Load a 16x32 f16 A/B fragment from a row-major [rows][ld] matrix.
// Lane (idx = lane&15 selects M or N row, kg = lane>>4 selects K half):
//   halves 0..7  <- K = k + kg*8 + {0..7}      (contiguous 16B)
//   halves 8..15 <- K = k + 16 + kg*8 + {0..7} (contiguous 16B)
// Matches the CDNA5 ISA 16-bit A-matrix (and symmetric B) VGPR layout.
__device__ __forceinline__ v16h load_frag(const _Float16* __restrict__ base,
                                          int ld, int row0, int k, int lane) {
  const int idx = lane & 15;
  const int kg  = lane >> 4;
  const _Float16* p = base + (long)(row0 + idx) * ld + k + kg * 8;
  v8h lo = *(const v8h*)(p);
  v8h hi = *(const v8h*)(p + 16);
  v16h r;
#pragma unroll
  for (int i = 0; i < 8; ++i) { r[i] = lo[i]; r[i + 8] = hi[i]; }
  return r;
}

__device__ __forceinline__ v8f wmma32(v16h a, v16h b, v8f c) {
  return __builtin_amdgcn_wmma_f32_16x16x32_f16(false, a, false, b, (short)0, c,
                                                false, false);
}

// ---------------------------------------------------------------------------
// f32 -> f16 conversion with optional column padding (zero fill).
// ---------------------------------------------------------------------------
__global__ void cvt_pad(_Float16* __restrict__ dst, const float* __restrict__ src,
                        int rows, int cols, int srcld, int dstld) {
  long total = (long)rows * dstld;
  for (long i = blockIdx.x * (long)blockDim.x + threadIdx.x; i < total;
       i += (long)gridDim.x * blockDim.x) {
    long r = i / dstld;
    int  c = (int)(i - r * dstld);
    dst[r * dstld + c] = (c < cols) ? (_Float16)src[r * srcld + c] : (_Float16)0.f;
  }
}

// ---------------------------------------------------------------------------
// C[M,N] (f32) = A16[M,K] * B16[N,K]^T (+ bias0 + bias1).
// Each wave computes a 16x64 strip (4 N-tiles) so the A fragment is reused 4x.
// Requires N % 64 == 0.
// ---------------------------------------------------------------------------
__global__ void gemm_f16_wmma(const _Float16* __restrict__ A,
                              const _Float16* __restrict__ Bm,
                              float* __restrict__ C, int M, int N, int K,
                              int lda, int ldb, int ldc,
                              const float* __restrict__ bias0,
                              const float* __restrict__ bias1) {
  const int lane = threadIdx.x & 31;
  const int wid  = blockIdx.x * (blockDim.x >> 5) + (threadIdx.x >> 5);
  const int ntN4 = N >> 6;                  // groups of 4 N-tiles
  const int strips = (M >> 4) * ntN4;
  if (wid >= strips) return;                // wave-uniform: EXEC stays all-ones
  const int tm  = wid / ntN4;
  const int tn0 = (wid % ntN4) << 2;
  v8f acc[4] = {{}, {}, {}, {}};
  for (int k = 0; k < K; k += 32) {
    v16h a = load_frag(A, lda, tm << 4, k, lane);
#pragma unroll
    for (int tt = 0; tt < 4; ++tt) {
      v16h b = load_frag(Bm, ldb, (tn0 + tt) << 4, k, lane);
      acc[tt] = wmma32(a, b, acc[tt]);
    }
  }
  const int kg = lane >> 4;
#pragma unroll
  for (int tt = 0; tt < 4; ++tt) {
    const int n = ((tn0 + tt) << 4) + (lane & 15);
    float bb = 0.f;
    if (bias0) bb += bias0[n];
    if (bias1) bb += bias1[n];
#pragma unroll
    for (int r = 0; r < 8; ++r) {
      const int m = (tm << 4) + r + (kg << 3);  // C/D layout: M = r + 8*(lane>>4)
      C[(long)m * ldc + n] = acc[tt][r] + bb;
    }
  }
}

// Shared helper: one wave computes its 8 tiles of a 64x(16*64) gate GEMM
// G16[64][1024] = A16[64][K] @ W16[1024][K]^T, as 2 passes of 4 N-tiles.
__device__ __forceinline__ void gate_gemm_8tiles(const _Float16* __restrict__ A16,
                                                 const _Float16* __restrict__ W16,
                                                 _Float16* __restrict__ G16,
                                                 int K, int wv, int lane) {
  const int tm  = wv >> 3;                  // 4 M-tiles
  const int kg  = lane >> 4;
#pragma unroll 1
  for (int half = 0; half < 2; ++half) {
    const int tn0 = ((wv & 7) << 3) + (half << 2);
    v8f acc[4] = {{}, {}, {}, {}};
#pragma unroll 1
    for (int k = 0; k < K; k += 32) {
      v16h a = load_frag(A16, K, tm << 4, k, lane);
#pragma unroll
      for (int tt = 0; tt < 4; ++tt) {
        v16h b = load_frag(W16, K, (tn0 + tt) << 4, k, lane);
        acc[tt] = wmma32(a, b, acc[tt]);
      }
    }
#pragma unroll
    for (int tt = 0; tt < 4; ++tt) {
      const int n = ((tn0 + tt) << 4) + (lane & 15);
#pragma unroll
      for (int r = 0; r < 8; ++r)
        G16[((tm << 4) + r + (kg << 3)) * 1024 + n] = (_Float16)acc[tt][r];
    }
  }
}

// ---------------------------------------------------------------------------
// Persistent encoder LSTM scan: one block per direction (blockIdx.x = dir),
// 1024 threads = 32 waves. Per step: G16 = h16 @ Whh^T via WMMA, then gates.
// Dynamic LDS: G16 (64*1024 f16 = 128KB) + h16 (64*256 f16 = 32KB).
// ---------------------------------------------------------------------------
__global__ __launch_bounds__(1024) void enc_scan(
    const float* __restrict__ P0, const float* __restrict__ P1, // [8192][1024]
    const _Float16* __restrict__ WhhF, const _Float16* __restrict__ WhhR, // [1024][256]
    const float* __restrict__ h0base, const float* __restrict__ c0base,   // [2][64][256]
    float* __restrict__ out) {                                  // [B][S][512]
  extern __shared__ char smem[];
  _Float16* G16 = (_Float16*)smem;                     // 64*1024 halves
  _Float16* h16 = (_Float16*)(smem + 64 * 1024 * 2);   // 64*256 halves

  const int dir = blockIdx.x;
  const float*     P   = dir ? P1 : P0;
  const _Float16*  Whh = dir ? WhhR : WhhF;
  const float*     h0  = h0base + dir * (B_ * H_);
  const float*     c0  = c0base + dir * (B_ * H_);
  const int colofs = dir * H_;

  const int tid  = threadIdx.x;
  const int lane = tid & 31, wv = tid >> 5;
  const int brow = tid >> 4;            // batch row 0..63
  const int cb   = (tid & 15) << 4;     // 16 contiguous hidden cols

  float creg[16];
#pragma unroll
  for (int j = 0; j < 16; ++j) {
    const int col = cb + j;
    creg[j] = c0[brow * H_ + col];
    h16[brow * H_ + col] = (_Float16)h0[brow * H_ + col];
  }
  __syncthreads();

  for (int t = 0; t < S_; ++t) {
    const int s = dir ? (S_ - 1 - t) : t;
    // --- G16 = h16[64,256] @ Whh^T[256,1024] ---
    gate_gemm_8tiles(h16, Whh, G16, H_, wv, lane);
    __syncthreads();
    // --- gates: g = G16 + P[b*S+s]; c,h update; h -> LDS(f16) + global(f32) ---
    const long mrow = (long)brow * S_ + s;
#pragma unroll 1
    for (int j = 0; j < 16; ++j) {
      const int col = cb + j;
      const float gi = (float)G16[brow * 1024 + col]       + P[mrow * 1024 + col];
      const float gf = (float)G16[brow * 1024 + col + 256] + P[mrow * 1024 + col + 256];
      const float gg = (float)G16[brow * 1024 + col + 512] + P[mrow * 1024 + col + 512];
      const float go = (float)G16[brow * 1024 + col + 768] + P[mrow * 1024 + col + 768];
      const float c  = sigf(gf) * creg[j] + sigf(gi) * tanhfast(gg);
      creg[j] = c;
      const float h = sigf(go) * tanhfast(c);
      out[mrow * 512 + colofs + col] = h;
      h16[brow * H_ + col] = (_Float16)h;
    }
    __syncthreads();
  }
}

// ---------------------------------------------------------------------------
// Persistent decoder: single 32-wave workgroup, 128 sequential steps.
// LDS layout (bytes): G16[0,131072) (aliased as f32 hproj), hid16[131072),
// score[163840), din16[196608), h1_16[266240), y[299008). Total 299264 B.
// ---------------------------------------------------------------------------
__global__ __launch_bounds__(1024) void dec_scan(
    const _Float16* __restrict__ Wh16,    // [256][256]
    const float* __restrict__ encproj,    // [8192][256] (includes attn_b)
    const float* __restrict__ encout,     // [8192][512]
    const float* __restrict__ vvec,       // [256]
    const _Float16* __restrict__ dWih0,   // [1024][544] (K padded 513->544)
    const float* __restrict__ Rl0,        // [64][1024] = dec_h0[0]@Whh0^T + biases
    const float* __restrict__ c0l0,       // [64][256]
    const _Float16* __restrict__ dWih1,   // [1024][256]
    const float* __restrict__ Rl1,        // [64][1024]
    const float* __restrict__ c0l1,       // [64][256]
    const float* __restrict__ linW,       // [768]
    const float* __restrict__ linb,       // [1]
    const float* __restrict__ dh0,        // [64][256] = dec_h0[0]
    float* __restrict__ outy) {           // [64][128]
  extern __shared__ char smem[];
  _Float16* G16    = (_Float16*)smem;                 // 128 KB
  float*    hproj  = (float*)smem;                    // alias (64 KB used)
  _Float16* hid16  = (_Float16*)(smem + 131072);      // 32 KB
  float*    scoreL = (float*)(smem + 163840);         // 32 KB
  _Float16* din16  = (_Float16*)(smem + 196608);      // 68 KB ([64][544])
  _Float16* h116   = (_Float16*)(smem + 266240);      // 32 KB
  float*    yL     = (float*)(smem + 299008);         // 256 B

  const int tid  = threadIdx.x;
  const int lane = tid & 31, wv = tid >> 5;
  const int brow = tid >> 4;
  const int cb   = (tid & 15) << 4;

#pragma unroll
  for (int q = 0; q < 16; ++q) {
    const int e = tid + (q << 10);          // 64*256 init elements
    hid16[e] = (_Float16)dh0[e];
  }
  if (tid < 64) yL[tid] = 0.f;
  __syncthreads();

  for (int t = 0; t < S_; ++t) {
    // --- 1) hproj = hidden @ Wh^T : 64 tiles, 2 per wave (shared A frag) ---
    {
      const int tm  = wv >> 3;
      const int tn0 = (wv & 7) << 1;
      v8f acc0 = {}, acc1 = {};
#pragma unroll 1
      for (int k = 0; k < D_; k += 32) {
        v16h a  = load_frag(hid16, D_, tm << 4, k, lane);
        v16h b0 = load_frag(Wh16, D_, tn0 << 4, k, lane);
        v16h b1 = load_frag(Wh16, D_, (tn0 + 1) << 4, k, lane);
        acc0 = wmma32(a, b0, acc0);
        acc1 = wmma32(a, b1, acc1);
      }
      const int kg = lane >> 4;
#pragma unroll
      for (int r = 0; r < 8; ++r) {
        const int m = (tm << 4) + r + (kg << 3);
        hproj[m * 256 + (tn0 << 4) + (lane & 15)]       = acc0[r];
        hproj[m * 256 + ((tn0 + 1) << 4) + (lane & 15)] = acc1[r];
      }
    }
    __syncthreads();
    // --- 2) score[b,s] = v . tanh(hproj[b] + encproj[b,s]) ---
#pragma unroll 1
    for (int q = 0; q < 8; ++q) {
      const int idx = tid + (q << 10);      // 8192 (b,s) pairs
      const int b = idx >> 7;
      const float* ep = encproj + (long)idx * 256;
      const float* hp = hproj + b * 256;
      __builtin_prefetch(ep + 256, 0, 1);   // next row stays hot in L2/L0
      float sc = 0.f;
#pragma unroll 1
      for (int d = 0; d < 256; ++d) sc += vvec[d] * tanhfast(hp[d] + ep[d]);
      scoreL[idx] = sc;
    }
    __syncthreads();
    // --- 3) softmax over s (one thread per batch row) ---
    if (tid < 64) {
      float mx = -3.4e38f;
      for (int s = 0; s < 128; ++s) mx = fmaxf(mx, scoreL[tid * 128 + s]);
      float sum = 0.f;
      for (int s = 0; s < 128; ++s) {
        const float e = __expf(scoreL[tid * 128 + s] - mx);
        scoreL[tid * 128 + s] = e;
        sum += e;
      }
      const float inv = 1.f / sum;
      for (int s = 0; s < 128; ++s) scoreL[tid * 128 + s] *= inv;
    }
    __syncthreads();
    // --- 4) context -> din16 = [y | context | pad] in f16 ---
#pragma unroll 1
    for (int q = 0; q < 32; ++q) {
      const int idx = tid + (q << 10);      // 64*512 outputs
      const int b = idx >> 9, f = idx & 511;
      const float* eo = encout + ((long)b * 128) * 512 + f;
      const float* aw = scoreL + b * 128;
      float cx = 0.f;
#pragma unroll 1
      for (int s = 0; s < 128; ++s) cx += aw[s] * eo[(long)s * 512];
      din16[b * 544 + 1 + f] = (_Float16)cx;
    }
    if (tid < 64) {
      din16[tid * 544] = (_Float16)yL[tid];
      for (int c = 513; c < 544; ++c) din16[tid * 544 + c] = (_Float16)0.f;
    }
    __syncthreads();
    // --- 5) G16 = din16 @ dWih0^T (K=544), then cell 0 -> h1 ---
    gate_gemm_8tiles(din16, dWih0, G16, 544, wv, lane);
    __syncthreads();
#pragma unroll 1
    for (int j = 0; j < 16; ++j) {
      const int col = cb + j;
      const float gi = (float)G16[brow * 1024 + col]       + Rl0[brow * 1024 + col];
      const float gf = (float)G16[brow * 1024 + col + 256] + Rl0[brow * 1024 + col + 256];
      const float gg = (float)G16[brow * 1024 + col + 512] + Rl0[brow * 1024 + col + 512];
      const float go = (float)G16[brow * 1024 + col + 768] + Rl0[brow * 1024 + col + 768];
      const float c1 = sigf(gf) * c0l0[brow * 256 + col] + sigf(gi) * tanhfast(gg);
      h116[brow * 256 + col] = (_Float16)(sigf(go) * tanhfast(c1));
    }
    __syncthreads();
    // --- 6) G16 = h1 @ dWih1^T (K=256), then cell 1 -> hidden(h2) ---
    gate_gemm_8tiles(h116, dWih1, G16, 256, wv, lane);
    __syncthreads();
#pragma unroll 1
    for (int j = 0; j < 16; ++j) {
      const int col = cb + j;
      const float gi = (float)G16[brow * 1024 + col]       + Rl1[brow * 1024 + col];
      const float gf = (float)G16[brow * 1024 + col + 256] + Rl1[brow * 1024 + col + 256];
      const float gg = (float)G16[brow * 1024 + col + 512] + Rl1[brow * 1024 + col + 512];
      const float go = (float)G16[brow * 1024 + col + 768] + Rl1[brow * 1024 + col + 768];
      const float c2 = sigf(gf) * c0l1[brow * 256 + col] + sigf(gi) * tanhfast(gg);
      hid16[brow * 256 + col] = (_Float16)(sigf(go) * tanhfast(c2));
    }
    __syncthreads();
    // --- 7) y = [h2 | context] @ linW^T + linb ---
    if (tid < 64) {
      float acc = linb[0];
      const _Float16* h2p = hid16 + tid * 256;
      for (int d = 0; d < 256; ++d) acc += (float)h2p[d] * linW[d];
      const _Float16* cxp = din16 + tid * 544 + 1;
      for (int f2 = 0; f2 < 512; ++f2) acc += (float)cxp[f2] * linW[256 + f2];
      yL[tid] = acc;
      outy[tid * 128 + t] = acc;
    }
    __syncthreads();
  }
}

// ---------------------------------------------------------------------------
// Host launcher
// ---------------------------------------------------------------------------
extern "C" void kernel_launch(void* const* d_in, const int* in_sizes, int n_in,
                              void* d_out, int out_size, void* d_ws, size_t ws_size,
                              hipStream_t stream) {
  (void)in_sizes; (void)out_size; (void)ws_size;
  const float* p[34];
  for (int i = 0; i < 34 && i < n_in; ++i) p[i] = (const float*)d_in[i];
  const float* x      = p[0];
  const float* enc_h0 = p[1];
  const float* enc_c0 = p[2];
  const float* dec_h0 = p[3];
  const float* dec_c0 = p[4];

  char* ws = (char*)d_ws;
  size_t off = 0;
  auto alloc = [&](size_t bytes) -> char* {
    char* r = ws + off;
    off += (bytes + 255) & ~(size_t)255;
    return r;
  };
  // f16 weights / activations
  char* X16   = alloc((size_t)8192 * 64 * 2);
  char* WIH0F = alloc((size_t)1024 * 64 * 2);
  char* WIH0R = alloc((size_t)1024 * 64 * 2);
  char* WHH0F = alloc((size_t)1024 * 256 * 2);
  char* WHH0R = alloc((size_t)1024 * 256 * 2);
  char* WIH1F = alloc((size_t)1024 * 512 * 2);
  char* WIH1R = alloc((size_t)1024 * 512 * 2);
  char* WHH1F = alloc((size_t)1024 * 256 * 2);
  char* WHH1R = alloc((size_t)1024 * 256 * 2);
  char* DWIH0 = alloc((size_t)1024 * 544 * 2);
  char* DWHH0 = alloc((size_t)1024 * 256 * 2);
  char* DWIH1 = alloc((size_t)1024 * 256 * 2);
  char* DWHH1 = alloc((size_t)1024 * 256 * 2);
  char* WH16  = alloc((size_t)256 * 256 * 2);
  char* WE16  = alloc((size_t)256 * 512 * 2);
  char* DH016 = alloc((size_t)2 * 64 * 256 * 2);
  // f32 buffers
  char* OUT0   = alloc((size_t)8192 * 512 * 4);
  char* OUT016 = alloc((size_t)8192 * 512 * 2);
  char* ENCOUT = alloc((size_t)8192 * 512 * 4);
  char* ENCO16 = alloc((size_t)8192 * 512 * 2);
  char* EPROJ  = alloc((size_t)8192 * 256 * 4);
  char* P0B    = alloc((size_t)8192 * 1024 * 4);
  char* P1B    = alloc((size_t)8192 * 1024 * 4);
  char* RL0    = alloc((size_t)64 * 1024 * 4);
  char* RL1    = alloc((size_t)64 * 1024 * 4);

  auto cvt = [&](char* dst, const float* src, int rows, int cols, int srcld, int dstld) {
    long total = (long)rows * dstld;
    int blocks = (int)((total + 255) / 256);
    if (blocks > 4096) blocks = 4096;
    hipLaunchKernelGGL(cvt_pad, dim3(blocks), dim3(256), 0, stream,
                       (_Float16*)dst, src, rows, cols, srcld, dstld);
  };
  auto gemm = [&](const char* A, const char* Bm, char* C, int M, int N, int K,
                  int lda, int ldb, int ldc, const float* b0, const float* b1) {
    const int strips = (M / 16) * (N / 64);   // 4 N-tiles per wave
    const int blocks = (strips + 7) / 8;
    hipLaunchKernelGGL(gemm_f16_wmma, dim3(blocks), dim3(256), 0, stream,
                       (const _Float16*)A, (const _Float16*)Bm, (float*)C,
                       M, N, K, lda, ldb, ldc, b0, b1);
  };

  const int ENC_SMEM = 64 * 1024 * 2 + 64 * 256 * 2;   // 163840 B
  const int DEC_SMEM = 299264;                          // see dec_scan layout
  hipFuncSetAttribute(reinterpret_cast<const void*>(enc_scan),
                      hipFuncAttributeMaxDynamicSharedMemorySize, ENC_SMEM);
  hipFuncSetAttribute(reinterpret_cast<const void*>(dec_scan),
                      hipFuncAttributeMaxDynamicSharedMemorySize, DEC_SMEM);

  // ---- conversions to f16 ----
  cvt(X16,   x,      8192, 64,  64,  64);
  cvt(WIH0F, p[5],   1024, 64,  64,  64);
  cvt(WHH0F, p[6],   1024, 256, 256, 256);
  cvt(WIH0R, p[9],   1024, 64,  64,  64);
  cvt(WHH0R, p[10],  1024, 256, 256, 256);
  cvt(WIH1F, p[13],  1024, 512, 512, 512);
  cvt(WHH1F, p[14],  1024, 256, 256, 256);
  cvt(WIH1R, p[17],  1024, 512, 512, 512);
  cvt(WHH1R, p[18],  1024, 256, 256, 256);
  cvt(DWIH0, p[21],  1024, 513, 513, 544);   // K padded 513 -> 544
  cvt(DWHH0, p[22],  1024, 256, 256, 256);
  cvt(DWIH1, p[25],  1024, 256, 256, 256);
  cvt(DWHH1, p[26],  1024, 256, 256, 256);
  cvt(WH16,  p[29],        256, 256, 768, 256);   // attn_W[:, :256]
  cvt(WE16,  p[29] + 256,  256, 512, 768, 512);   // attn_W[:, 256:]
  cvt(DH016, dec_h0, 128, 256, 256, 256);

  // ---- encoder layer 0 ----
  gemm(X16, WIH0F, P0B, 8192, 1024, 64, 64, 64, 1024, p[7],  p[8]);
  gemm(X16, WIH0R, P1B, 8192, 1024, 64, 64, 64, 1024, p[11], p[12]);
  hipLaunchKernelGGL(enc_scan, dim3(2), dim3(1024), ENC_SMEM, stream,
                     (const float*)P0B, (const float*)P1B,
                     (const _Float16*)WHH0F, (const _Float16*)WHH0R,
                     enc_h0, enc_c0, (float*)OUT0);
  cvt(OUT016, (const float*)OUT0, 8192, 512, 512, 512);

  // ---- encoder layer 1 ----
  gemm(OUT016, WIH1F, P0B, 8192, 1024, 512, 512, 512, 1024, p[15], p[16]);
  gemm(OUT016, WIH1R, P1B, 8192, 1024, 512, 512, 512, 1024, p[19], p[20]);
  hipLaunchKernelGGL(enc_scan, dim3(2), dim3(1024), ENC_SMEM, stream,
                     (const float*)P0B, (const float*)P1B,
                     (const _Float16*)WHH1F, (const _Float16*)WHH1R,
                     enc_h0 + 2 * 64 * 256, enc_c0 + 2 * 64 * 256, (float*)ENCOUT);
  cvt(ENCO16, (const float*)ENCOUT, 8192, 512, 512, 512);

  // ---- attention precompute: enc_proj = enc_out @ We^T + attn_b ----
  gemm(ENCO16, WE16, EPROJ, 8192, 256, 512, 512, 512, 256, p[30], nullptr);

  // ---- decoder step-invariant recurrent terms (state resets every step) ----
  gemm(DH016,                 DWHH0, RL0, 64, 1024, 256, 256, 256, 1024, p[23], p[24]);
  gemm(DH016 + 64 * 256 * 2,  DWHH1, RL1, 64, 1024, 256, 256, 256, 1024, p[27], p[28]);

  // ---- decoder scan ----
  hipLaunchKernelGGL(dec_scan, dim3(1), dim3(1024), DEC_SMEM, stream,
                     (const _Float16*)WH16, (const float*)EPROJ,
                     (const float*)ENCOUT, p[31],
                     (const _Float16*)DWIH0, (const float*)RL0, dec_c0,
                     (const _Float16*)DWIH1, (const float*)RL1, dec_c0 + 64 * 256,
                     p[32], p[33], dec_h0, (float*)d_out);
}
